// MultiHeadImageAttentionBlock_59399397703974
// MI455X (gfx1250) — matrix operand order repaired and checked by reading, not compiled, not run on hardware
//
#include <hip/hip_runtime.h>
#include <hip/hip_bf16.h>

typedef __attribute__((ext_vector_type(16))) _Float16 v16h;
typedef __attribute__((ext_vector_type(8)))  _Float16 v8h;
typedef __attribute__((ext_vector_type(8)))  float    v8f;
typedef __attribute__((ext_vector_type(4)))  float    v4f;
typedef __attribute__((ext_vector_type(4)))  unsigned int u32x4;
typedef __attribute__((ext_vector_type(8)))  int      i32x8;
typedef __attribute__((ext_vector_type(4)))  int      i32x4;

#define HW    1024      // 32*32 pixels
#define NHEAD 8
#define DK    32
#define NSEQ  4
#define BATCH 4
#define SKV   4096      // NSEQ*HW
#define PADW  34        // 32 + 2 halo columns (zeroed)
#define VCHUNK 512      // pixels of V staged in LDS per TDM transfer

// --- WMMA 16x16x32 f16 operand index maps used by this code (contiguous k order, not ISA table order) ---
// A (16x32): lane l=lane&15 -> row M=l; hi=lane>>4.
//   element i: K = i + 16*hi   (one contiguous 16-run)
// B (32x16): lane l -> col N=l; element i: K = i + 16*hi (one contiguous 16-run)
// C/D (16x16 f32): element r: M = r + 8*hi, N = lane&15
__device__ __forceinline__ int a_kidx(int i, int hi) {
  return i + (hi << 4);
}

// ---------------------------------------------------------------------
// Swizzle conv weights (Cout,Cin,3,3) fp32 -> per-lane WMMA A-fragment order:
// out[tid] with tid = (((rs*CoutT+ct)*Ksteps+ks)<<9) + lane*16 + i
__global__ void swizzle_weights_kernel(const float* __restrict__ w,
                                       _Float16* __restrict__ out,
                                       int Cin, int Cout) {
  int tid = blockIdx.x * 256 + threadIdx.x;
  int total = 9 * Cout * Cin;
  if (tid >= total) return;
  int i    = tid & 15;
  int lane = (tid >> 4) & 31;
  int rest = tid >> 9;
  int Ksteps = Cin >> 5, CoutT = Cout >> 4;
  int ks = rest % Ksteps; rest /= Ksteps;
  int ct = rest % CoutT;  int rs = rest / CoutT;
  int hi = lane >> 4, l = lane & 15;
  int cout = ct * 16 + l;
  int ci = (ks << 5) + a_kidx(i, hi);
  out[tid] = (_Float16)w[((size_t)cout * Cin + ci) * 9 + rs];
}

// ---------------------------------------------------------------------
// NCHW fp32 -> pixel-major f16 with zero-padded x halo: out[img][y][x+1][ci]
__global__ void to_pixmajor_pad_kernel(const float* __restrict__ in,
                                       _Float16* __restrict__ out, int Cin) {
  int p  = blockIdx.x * 256 + threadIdx.x;   // grid.x = HW/256
  int ci = blockIdx.y;
  int img = blockIdx.z;
  float v = in[((size_t)img * Cin + ci) * HW + p];
  int y = p >> 5, x = p & 31;
  out[((size_t)img * 32 * PADW + y * PADW + x + 1) * Cin + ci] = (_Float16)v;
}

// ---------------------------------------------------------------------
// 3x3 SAME conv as 9 shifted GEMMs, all-WMMA, no division, no exec-diverged
// loads. inT: padded pixel-major f16. wSwz: lane-swizzled f16 weights.
__global__ void conv3x3_wmma_kernel(const _Float16* __restrict__ inT,
                                    const _Float16* __restrict__ wSwz,
                                    const float* __restrict__ bias,
                                    float* __restrict__ out32,
                                    _Float16* __restrict__ outPM,
                                    _Float16* __restrict__ outCM,
                                    int Cin, int Cout) {
  const int lane = threadIdx.x & 31;
  const int wave = threadIdx.x >> 5;
  const int hi = lane >> 4, l = lane & 15;
  const int ptile = blockIdx.x * 4 + wave;   // 64 pixel tiles of 16
  const int p0 = ptile * 16;
  const int py = p0 >> 5, px0 = p0 & 31;     // 16 consecutive x, same row
  const int ct = blockIdx.y;                 // cout tile
  const int cout0 = ct * 16;
  const int img = blockIdx.z;
  const int Ksteps = Cin >> 5;
  const int CoutT = Cout >> 4;
  const _Float16* inImg = inT + (size_t)img * 32 * PADW * Cin;

  v8f acc = {};
#pragma unroll
  for (int r = 0; r < 3; ++r) {
    const int yy = py + r - 1;
    if (yy < 0 || yy >= 32) continue;        // uniform: zero-pad rows skipped
    const _Float16* rowB = inImg + (size_t)yy * PADW * Cin;
#pragma unroll
    for (int s = 0; s < 3; ++s) {
      const _Float16* bptr = rowB + (size_t)(px0 + l + s) * Cin + (hi << 4);
      const _Float16* aptr = wSwz +
          ((((size_t)(r * 3 + s) * CoutT + ct) * Ksteps) << 9) + lane * 16;
      for (int ks = 0; ks < Ksteps; ++ks) {
        v16h a = *(const v16h*)(aptr + ((size_t)ks << 9));   // 32B contiguous
        v16h b = *(const v16h*)(bptr + (ks << 5));           // 16 consecutive ci
        acc = __builtin_amdgcn_wmma_f32_16x16x32_f16(false, a, false, b,
                                                     (short)0, acc, false, false);
      }
    }
  }

  float bsum[8];
#pragma unroll
  for (int r = 0; r < 8; ++r) bsum[r] = acc[r] + bias[cout0 + r + (hi << 3)];

  if (outPM) {   // pixel-major f16: [p][ch]; lane's 8 channels contiguous
    v8h hv;
#pragma unroll
    for (int r = 0; r < 8; ++r) hv[r] = (_Float16)bsum[r];
    *(v8h*)(outPM + ((size_t)img * HW + p0 + l) * Cout + cout0 + (hi << 3)) = hv;
  }
  if (outCM) {   // channel-major f16: [ch][p]
#pragma unroll
    for (int r = 0; r < 8; ++r) {
      int co = cout0 + r + (hi << 3);
      outCM[((size_t)img * Cout + co) * HW + p0 + l] = (_Float16)bsum[r];
    }
  }
  if (out32) {   // NCHW fp32
#pragma unroll
    for (int r = 0; r < 8; ++r) {
      int co = cout0 + r + (hi << 3);
      out32[((size_t)img * Cout + co) * HW + p0 + l] = bsum[r];
    }
  }
}

// ---------------------------------------------------------------------
// scores[b,h,q,k] = (1/sqrt(32)) * sum_d Q[q,d] * K[k,d]; one WMMA per tile.
__global__ void scores_wmma_kernel(const _Float16* __restrict__ Qt,
                                   const _Float16* __restrict__ Kt,
                                   float* __restrict__ scores) {
  const int lane = threadIdx.x & 31;
  const int wave = threadIdx.x >> 5;
  const int hi = lane >> 4, l = lane & 15;
  const int k0 = (blockIdx.x * 4 + wave) * 16;
  const int q0 = blockIdx.y * 16;
  const int bh = blockIdx.z;
  const int b = bh >> 3, h = bh & 7;

  const _Float16* qrow = Qt + ((size_t)b * HW + q0 + l) * 256 + h * DK;
  v8h a0 = *(const v8h*)(qrow + (hi << 4));
  v8h a1 = *(const v8h*)(qrow + 8 + (hi << 4));
  v16h a;
#pragma unroll
  for (int j = 0; j < 8; ++j) { a[j] = a0[j]; a[8 + j] = a1[j]; }

  const int kcol = k0 + l;
  const int nimg = kcol >> 10, p = kcol & 1023;
  v16h bm = *(const v16h*)(Kt + (((size_t)nimg * BATCH + b) * HW + p) * 256 +
                           h * DK + (hi << 4));
  v8f acc = {};
  acc = __builtin_amdgcn_wmma_f32_16x16x32_f16(false, a, false, bm,
                                               (short)0, acc, false, false);
  const float sc = 0.17677669529663687f;  // 1/sqrt(32)
#pragma unroll
  for (int r = 0; r < 8; ++r) {
    int q = q0 + r + (hi << 3);
    // attn stream is 512MB >> L2: non-temporal store
    __builtin_nontemporal_store(acc[r] * sc,
                                &scores[((size_t)bh * HW + q) * SKV + k0 + l]);
  }
}

// ---------------------------------------------------------------------
// Softmax over the QUERY axis (dim=-2): one thread per column k, online
// max+sum pass then normalize in place; coalesced, fully non-temporal.
__global__ void softmax_q_kernel(float* __restrict__ attn) {
  const int k = blockIdx.x * 256 + threadIdx.x;
  const int bh = blockIdx.y;
  float* col = attn + (size_t)bh * HW * SKV + k;
  float m = -3.0e38f, s = 0.f;
  for (int q = 0; q < HW; ++q) {
    float v = __builtin_nontemporal_load(&col[(size_t)q * SKV]);
    if (v > m) { s *= __expf(m - v); m = v; }
    s += __expf(v - m);
  }
  const float inv = 1.f / s;
  for (int q = 0; q < HW; ++q) {
    float v = __builtin_nontemporal_load(&col[(size_t)q * SKV]);
    __builtin_nontemporal_store(__expf(v - m) * inv, &col[(size_t)q * SKV]);
  }
}

// ---------------------------------------------------------------------
// X[q, h*32+dv] = sum_k attn[b,h,q,k] * V[b,h,dv,k]  (K-dim = 4096)
// V slice (shared by all 4 waves of the block) is staged into LDS in
// 16x512 f16 chunks by the Tensor Data Mover; B-operand reads become
// ds_load_b128. attn A-loads are non-temporal fp32 vector loads.
__global__ void attnv_wmma_kernel(const float* __restrict__ attn,
                                  const _Float16* __restrict__ Vh,
                                  _Float16* __restrict__ Xp) {
  const int lane = threadIdx.x & 31;
  const int wave = threadIdx.x >> 5;
  const int hi = lane >> 4, l = lane & 15;
  const int dv0 = blockIdx.x * 16;                // 2 tiles over d_v
  const int q0 = (blockIdx.y * 4 + wave) * 16;    // 64 query tiles
  const int bh = blockIdx.z;
  const int b = bh >> 3, h = bh & 7;
  const float* arow = attn + ((size_t)bh * HW + q0 + l) * SKV;

  __shared__ _Float16 vs[16 * VCHUNK];            // 16KB V staging tile

  v8f acc = {};
  for (int c0 = 0; c0 < SKV; c0 += VCHUNK) {
    const int nimg = c0 >> 10, p0c = c0 & 1023;   // chunk within one image
    const _Float16* gsrc =
        Vh + (((size_t)nimg * BATCH + b) * 256 + h * DK + dv0) * HW + p0c;
    if (wave == 0) {
#if __has_builtin(__builtin_amdgcn_tensor_load_to_lds)
      // TDM 2D descriptor: 2B elements, tile 512x16, row stride 1024 elems
      unsigned long long ga = (unsigned long long)(uintptr_t)gsrc;
      u32x4 g0;
      g0[0] = 1u;                                      // count=1 (valid D#)
      g0[1] = (unsigned)(uintptr_t)vs;                 // LDS byte address
      g0[2] = (unsigned)(ga & 0xFFFFFFFFu);            // global_addr[31:0]
      g0[3] = (unsigned)((ga >> 32) & 0x01FFFFFFu) | (2u << 30);  // type=2
      i32x8 g1;
      g1[0] = (1 << 16);                               // data_size=1 -> 2 bytes
      g1[1] = (HW & 0xFFFF) << 16;                     // tensor_dim0 = 1024
      g1[2] = ((16 & 0xFFFF) << 16);                   // tensor_dim1 = 16
      g1[3] = ((VCHUNK & 0xFFFF) << 16);               // tile_dim0 = 512
      g1[4] = 16;                                      // tile_dim1 = 16
      g1[5] = HW;                                      // dim0_stride = 1024
      g1[6] = 0; g1[7] = 0;
      i32x4 g2 = {0, 0, 0, 0}, g3 = {0, 0, 0, 0};
#if __clang_major__ >= 23
      i32x8 g4 = {0, 0, 0, 0, 0, 0, 0, 0};
      __builtin_amdgcn_tensor_load_to_lds(g0, g1, g2, g3, g4, 0);
#else
      __builtin_amdgcn_tensor_load_to_lds(g0, g1, g2, g3, 0);
#endif
#if __has_builtin(__builtin_amdgcn_s_wait_tensorcnt)
      __builtin_amdgcn_s_wait_tensorcnt(0);
#else
      asm volatile("s_wait_tensorcnt 0x0" ::: "memory");
#endif
#else
      // fallback: cooperative synchronous copy by wave 0
      for (int j = lane; j < 1024; j += 32) {
        int row = j >> 6, c8 = (j & 63) << 3;
        *(v8h*)&vs[row * VCHUNK + c8] = *(const v8h*)(gsrc + (size_t)row * HW + c8);
      }
#endif
    }
    __syncthreads();

    for (int kk0 = c0; kk0 < c0 + VCHUNK; kk0 += 32) {
      __builtin_prefetch(arow + kk0 + 256, 0, 1);   // global_prefetch_b8
      // A: one contiguous 16-float run of attn -> f16 (non-temporal)
      const float* run0 = arow + kk0 + (hi << 4);
      const float* run1 = arow + kk0 + 8 + (hi << 4);
      v4f f0 = __builtin_nontemporal_load((const v4f*)run0);
      v4f f1 = __builtin_nontemporal_load((const v4f*)(run0 + 4));
      v4f f2 = __builtin_nontemporal_load((const v4f*)run1);
      v4f f3 = __builtin_nontemporal_load((const v4f*)(run1 + 4));
      v16h a;
#pragma unroll
      for (int j = 0; j < 4; ++j) {
        a[j]      = (_Float16)f0[j];
        a[4 + j]  = (_Float16)f1[j];
        a[8 + j]  = (_Float16)f2[j];
        a[12 + j] = (_Float16)f3[j];
      }
      // B from LDS: row = channel-in-tile (= lane&15), 16 consecutive pixels
      v16h bm = *(const v16h*)&vs[l * VCHUNK + (kk0 - c0) + (hi << 4)];
      acc = __builtin_amdgcn_wmma_f32_16x16x32_f16(false, a, false, bm,
                                                   (short)0, acc, false, false);
    }
    __syncthreads();
  }
#pragma unroll
  for (int r = 0; r < 8; ++r) {
    int q = q0 + r + (hi << 3);
    int y = q >> 5, x = q & 31;
    Xp[((size_t)b * 32 * PADW + y * PADW + x + 1) * 256 + h * DK + dv0 + l] =
        (_Float16)acc[r];
  }
}

// ---------------------------------------------------------------------
// y = LN(conv_o + 2*query) * ln_w + ln_b  (per-batch mean/var, eps=1e-6)
__global__ void resid_ln_kernel(const float* __restrict__ convo,
                                const float* __restrict__ query,
                                const float* __restrict__ lnw,
                                const float* __restrict__ lnb,
                                float* __restrict__ y) {
  const int b = blockIdx.x;
  const int tid = threadIdx.x;            // 256 threads
  const int N = 64 * HW;
  const float* cb = convo + (size_t)b * N;
  const float* qb = query + (size_t)b * N;
  __shared__ float s1[256], s2[256];
  float sum = 0.f, sq = 0.f;
  for (int i = tid; i < N; i += 256) {
    float v = cb[i] + 2.f * qb[i];
    sum += v; sq += v * v;
  }
  s1[tid] = sum; s2[tid] = sq;
  __syncthreads();
  for (int off = 128; off > 0; off >>= 1) {
    if (tid < off) { s1[tid] += s1[tid + off]; s2[tid] += s2[tid + off]; }
    __syncthreads();
  }
  const float mean = s1[0] / (float)N;
  const float var  = s2[0] / (float)N - mean * mean;
  const float rstd = rsqrtf(var + 1e-6f);
  for (int i = tid; i < N; i += 256) {
    float v = cb[i] + 2.f * qb[i];
    y[(size_t)b * N + i] = (v - mean) * rstd * lnw[i] + lnb[i];
  }
}

// ---------------------------------------------------------------------
extern "C" void kernel_launch(void* const* d_in, const int* in_sizes, int n_in,
                              void* d_out, int out_size, void* d_ws, size_t ws_size,
                              hipStream_t stream) {
  const float* query = (const float*)d_in[0];
  const float* seq   = (const float*)d_in[1];
  const float* wq = (const float*)d_in[2];
  const float* bq = (const float*)d_in[3];
  const float* wk = (const float*)d_in[4];
  const float* bk = (const float*)d_in[5];
  const float* wv = (const float*)d_in[6];
  const float* bv = (const float*)d_in[7];
  const float* wo = (const float*)d_in[8];
  const float* bo = (const float*)d_in[9];
  const float* lnw = (const float*)d_in[10];
  const float* lnb = (const float*)d_in[11];

  float* y_out    = (float*)d_out;
  float* attn_out = y_out + (size_t)BATCH * 64 * HW;  // (B,8,1024,4096) fp32

  char* w = (char*)d_ws;
  auto carve = [&](size_t bytes) {
    char* p = w; w += (bytes + 255) & ~(size_t)255; return p;
  };
  const int WN = 147456;  // 9*Cout*Cin for all four convs
  _Float16* wqS = (_Float16*)carve(WN * sizeof(_Float16));
  _Float16* wkS = (_Float16*)carve(WN * sizeof(_Float16));
  _Float16* wvS = (_Float16*)carve(WN * sizeof(_Float16));
  _Float16* woS = (_Float16*)carve(WN * sizeof(_Float16));
  const size_t INTQ = (size_t)BATCH * 32 * PADW * 64;
  const size_t INTS = (size_t)NSEQ * BATCH * 32 * PADW * 64;
  const size_t XPN  = (size_t)BATCH * 32 * PADW * 256;
  _Float16* inTq = (_Float16*)carve(INTQ * sizeof(_Float16));
  _Float16* inTs = (_Float16*)carve(INTS * sizeof(_Float16));
  _Float16* Xp   = (_Float16*)carve(XPN  * sizeof(_Float16));
  _Float16* Qt = (_Float16*)carve((size_t)BATCH      * HW * 256 * sizeof(_Float16));
  _Float16* Kt = (_Float16*)carve((size_t)NSEQ*BATCH * HW * 256 * sizeof(_Float16));
  _Float16* Vh = (_Float16*)carve((size_t)NSEQ*BATCH * HW * 256 * sizeof(_Float16));
  float*    Oc = (float*)   carve((size_t)BATCH * 64 * HW * sizeof(float));

  // 0) zero padded buffers (halo columns must be 0; capture-safe)
  hipMemsetAsync(inTq, 0, INTQ * sizeof(_Float16), stream);
  hipMemsetAsync(inTs, 0, INTS * sizeof(_Float16), stream);
  hipMemsetAsync(Xp,   0, XPN  * sizeof(_Float16), stream);

  // 1) weight swizzle into per-lane WMMA A-fragment order
  swizzle_weights_kernel<<<(WN + 255) / 256, 256, 0, stream>>>(wq, wqS, 64, 256);
  swizzle_weights_kernel<<<(WN + 255) / 256, 256, 0, stream>>>(wk, wkS, 64, 256);
  swizzle_weights_kernel<<<(WN + 255) / 256, 256, 0, stream>>>(wv, wvS, 64, 256);
  swizzle_weights_kernel<<<(WN + 255) / 256, 256, 0, stream>>>(wo, woS, 256, 64);

  // 2) inputs -> padded pixel-major f16
  to_pixmajor_pad_kernel<<<dim3(4, 64, BATCH),      256, 0, stream>>>(query, inTq, 64);
  to_pixmajor_pad_kernel<<<dim3(4, 64, NSEQ*BATCH), 256, 0, stream>>>(seq,   inTs, 64);

  dim3 blk(128);  // 4 waves (wave32)
  // 3) Q/K/V convs (WMMA); outputs in consumer-native f16 layouts
  conv3x3_wmma_kernel<<<dim3(16, 16, BATCH),      blk, 0, stream>>>(
      inTq, wqS, bq, nullptr, Qt, nullptr, 64, 256);
  conv3x3_wmma_kernel<<<dim3(16, 16, NSEQ*BATCH), blk, 0, stream>>>(
      inTs, wkS, bk, nullptr, Kt, nullptr, 64, 256);
  conv3x3_wmma_kernel<<<dim3(16, 16, NSEQ*BATCH), blk, 0, stream>>>(
      inTs, wvS, bv, nullptr, nullptr, Vh, 64, 256);
  // 4) scores -> attn region of d_out (one WMMA per 16x16 tile, NT stores)
  scores_wmma_kernel<<<dim3(64, 64, BATCH*NHEAD), blk, 0, stream>>>(Qt, Kt, attn_out);
  // 5) softmax over query axis, in place (2 NT passes over 512MB)
  softmax_q_kernel<<<dim3(SKV / 256, BATCH*NHEAD), dim3(256), 0, stream>>>(attn_out);
  // 6) attn @ V (TDM-staged V in LDS) -> padded pixel-major f16
  attnv_wmma_kernel<<<dim3(2, 16, BATCH*NHEAD), blk, 0, stream>>>(attn_out, Vh, Xp);
  // 7) output conv (Cin=256, Cout=64) -> fp32 NCHW
  conv3x3_wmma_kernel<<<dim3(16, 4, BATCH), blk, 0, stream>>>(
      Xp, woS, bo, Oc, nullptr, nullptr, 256, 64);
  // 8) double residual + LayerNorm -> y
  resid_ln_kernel<<<dim3(BATCH), dim3(256), 0, stream>>>(Oc, query, lnw, lnb, y_out);
}
